// FlatMemoryCell_13237089206561
// MI455X (gfx1250) — compile-verified, matrix operand order checked by reading, not run on hardware
//
#include <hip/hip_runtime.h>
#include <hip/hip_bf16.h>
#include <math.h>

typedef __attribute__((ext_vector_type(16))) __bf16 v16bf;
typedef __attribute__((ext_vector_type(8)))  __bf16 v8bf;
typedef __attribute__((ext_vector_type(8)))  float  v8f;

union Frag16 { v16bf v; v8bf h[2]; };

#define EPSF 1e-8f

static constexpr int BSZ = 4096;
static constexpr int H   = 512;
static constexpr int MM  = 900;
static constexpr int MH  = 1412;        // H + MM
static constexpr int KP  = 1472;        // MH padded to multiple of 64
static constexpr int NP  = 320;         // 272 (120+120+16+16) padded to multiple of 64

// ---------------------------------------------------------------------------
// Pack kernels: fp32 -> bf16 operand staging.
// B operands are packed N-major (BT[n][k]) so GEMM tile loads are contiguous.
// For nn.Linear weights ([out,in] row-major) this is the identity layout.
// ---------------------------------------------------------------------------
__global__ void pack_A1(const float* __restrict__ x, const float* __restrict__ hprev,
                        __bf16* __restrict__ A1) {
  int idx = blockIdx.x * 256 + threadIdx.x;          // BSZ*1024
  if (idx >= BSZ * 1024) return;
  int b = idx >> 10, c = idx & 1023;
  float v = (c < 512) ? x[b * 512 + c] : hprev[b * 512 + (c - 512)];
  A1[idx] = (__bf16)v;
}

__global__ void pack_BT1(const float* __restrict__ W_i2h, const float* __restrict__ W_h2h,
                         __bf16* __restrict__ BT1) {
  int idx = blockIdx.x * 256 + threadIdx.x;          // 512*1024, BT1[n][k] = W[k][n]
  if (idx >= 512 * 1024) return;
  int n = idx >> 10, k = idx & 1023;
  float v = (k < 512) ? W_i2h[k * 512 + n] : W_h2h[(k - 512) * 512 + n];
  BT1[idx] = (__bf16)v;
}

__global__ void pack_CM(const float* __restrict__ hint, const float* __restrict__ mem,
                        __bf16* __restrict__ CM) {
  int idx = blockIdx.x * 256 + threadIdx.x;          // BSZ*KP
  if (idx >= BSZ * KP) return;
  int b = idx / KP, c = idx - b * KP;
  float v = 0.f;
  if (c < 512)       v = hint[b * 512 + c];
  else if (c < MH)   v = mem[b * MM + (c - 512)];
  CM[idx] = (__bf16)v;
}

__global__ void pack_BT2(const float* __restrict__ W_va, const float* __restrict__ W_vb,
                         const float* __restrict__ W_a,  const float* __restrict__ W_b,
                         __bf16* __restrict__ BT2) {
  int idx = blockIdx.x * 256 + threadIdx.x;          // NP*KP, BT2[n][k] = W_sel[n][k]
  if (idx >= NP * KP) return;
  int n = idx / KP, k = idx - n * KP;
  float v = 0.f;
  if (k < MH) {
    if (n < 120)      v = W_va[n * MH + k];
    else if (n < 240) v = W_vb[(n - 120) * MH + k];
    else if (n < 256) v = W_a[(n - 240) * MH + k];
    else if (n < 272) v = W_b[(n - 256) * MH + k];
  }
  BT2[idx] = (__bf16)v;
}

__global__ void pack_BT3(const float* __restrict__ W_o, __bf16* __restrict__ BT3) {
  int idx = blockIdx.x * 256 + threadIdx.x;          // 512*KP, BT3[n][k] = W_o[n][k]
  if (idx >= 512 * KP) return;
  int n = idx / KP, k = idx - n * KP;
  BT3[idx] = (__bf16)((k < MH) ? W_o[n * MH + k] : 0.f);
}

__global__ void pack_bias2(const float* __restrict__ b_va, const float* __restrict__ b_vb,
                           const float* __restrict__ b_a,  const float* __restrict__ b_b,
                           float* __restrict__ bias2) {
  int n = blockIdx.x * 64 + threadIdx.x;
  if (n >= NP) return;
  float v = 0.f;
  if (n < 120)      v = b_va[n];
  else if (n < 240) v = b_vb[n - 120];
  else if (n < 256) v = b_a[n - 240];
  else if (n < 272) v = b_b[n - 256];
  bias2[n] = v;
}

// ---------------------------------------------------------------------------
// WMMA bf16 GEMM: C[M,N] = A[M,K] * BT[N,K]^T + bias[N]
// Block: 256 threads = 8 waves, tile 128(M) x 64(N); wave -> 32x32.
// K-step 64 (8 WMMAs/wave per step); register double-buffered global loads;
// global_prefetch of tile t+2. LDS rows padded to 72 halves (144B stride,
// conflict-free fragment reads, 16B-aligned v8bf accesses only).
// ---------------------------------------------------------------------------
template <bool TANH_DUAL>
__global__ __launch_bounds__(256)
void gemm_wmma_bf16(const __bf16* __restrict__ A, int lda,
                    const __bf16* __restrict__ BT, int ldbt,
                    const float* __restrict__ bias,
                    float* __restrict__ Cf, int ldc,
                    __bf16* __restrict__ Cb, int ldcb,
                    int K) {
  __shared__ __bf16 As[128][72];
  __shared__ __bf16 Bs[64][72];

  const int tid  = threadIdx.x;
  const int wave = tid >> 5;
  const int lane = tid & 31;
  const int wr   = wave >> 1;                    // 0..3  (M direction)
  const int wc   = wave & 1;                     // 0..1  (N direction)
  const int bm   = blockIdx.y * 128;
  const int bn   = blockIdx.x * 64;
  const int half = lane >> 4;                    // K half select
  const int l15  = lane & 15;

  // per-thread staging assignments
  const int ar = tid >> 1;                       // A row 0..127
  const int ac = (tid & 1) * 32;                 // A col 0 / 32 (32 halves = 4 x v8bf)
  const int br = tid >> 2;                       // B row 0..63
  const int bc = (tid & 3) * 16;                 // B col (16 halves = 2 x v8bf)
  const __bf16* gA = A  + (size_t)(bm + ar) * lda  + ac;
  const __bf16* gB = BT + (size_t)(bn + br) * ldbt + bc;

  v8bf ra[4], rb[2];
  auto loadG = [&](int k0) {
#pragma unroll
    for (int e = 0; e < 4; ++e) ra[e] = *(const v8bf*)(gA + k0 + e * 8);
#pragma unroll
    for (int e = 0; e < 2; ++e) rb[e] = *(const v8bf*)(gB + k0 + e * 8);
    if (k0 + 64 < K) {
      __builtin_prefetch(gA + k0 + 64, 0, 0);
      __builtin_prefetch(gB + k0 + 64, 0, 0);
    }
  };
  auto storeL = [&]() {
#pragma unroll
    for (int e = 0; e < 4; ++e) *(v8bf*)&As[ar][ac + e * 8] = ra[e];
#pragma unroll
    for (int e = 0; e < 2; ++e) *(v8bf*)&Bs[br][bc + e * 8] = rb[e];
  };

  v8f acc[2][2] = {};

  loadG(0);
  storeL();
  __syncthreads();

  for (int k0 = 0; k0 < K; k0 += 64) {
    const bool has_next = (k0 + 64) < K;
    if (has_next) loadG(k0 + 64);

#pragma unroll
    for (int kk = 0; kk < 64; kk += 32) {
      const int c = kk + half * 16;
      Frag16 a0, a1, b0, b1;
      a0.h[0] = *(const v8bf*)&As[wr * 32 +      l15][c];
      a0.h[1] = *(const v8bf*)&As[wr * 32 +      l15][c + 8];
      a1.h[0] = *(const v8bf*)&As[wr * 32 + 16 + l15][c];
      a1.h[1] = *(const v8bf*)&As[wr * 32 + 16 + l15][c + 8];
      b0.h[0] = *(const v8bf*)&Bs[wc * 32 +      l15][c];
      b0.h[1] = *(const v8bf*)&Bs[wc * 32 +      l15][c + 8];
      b1.h[0] = *(const v8bf*)&Bs[wc * 32 + 16 + l15][c];
      b1.h[1] = *(const v8bf*)&Bs[wc * 32 + 16 + l15][c + 8];

      acc[0][0] = __builtin_amdgcn_wmma_f32_16x16x32_bf16(false, a0.v, false, b0.v, (short)0, acc[0][0], false, false);
      acc[0][1] = __builtin_amdgcn_wmma_f32_16x16x32_bf16(false, a0.v, false, b1.v, (short)0, acc[0][1], false, false);
      acc[1][0] = __builtin_amdgcn_wmma_f32_16x16x32_bf16(false, a1.v, false, b0.v, (short)0, acc[1][0], false, false);
      acc[1][1] = __builtin_amdgcn_wmma_f32_16x16x32_bf16(false, a1.v, false, b1.v, (short)0, acc[1][1], false, false);
    }
    __syncthreads();
    if (has_next) {
      storeL();
      __syncthreads();
    }
  }

  // Epilogue: C/D layout -> lane l15 = N, vgpr r -> M = r + 8*half
#pragma unroll
  for (int mt = 0; mt < 2; ++mt)
#pragma unroll
    for (int nt = 0; nt < 2; ++nt) {
      int col = bn + wc * 32 + nt * 16 + l15;
      float bv = bias[col];
#pragma unroll
      for (int r = 0; r < 8; ++r) {
        int row = bm + wr * 32 + mt * 16 + half * 8 + r;
        float v = acc[mt][nt][r] + bv;
        if (TANH_DUAL) v = tanhf(v);
        Cf[(size_t)row * ldc + col] = v;
        if (TANH_DUAL) Cb[(size_t)row * ldcb + col] = (__bf16)v;
      }
    }
}

// ---------------------------------------------------------------------------
// Addressing kernel: one 512-thread block (16 waves) per batch row.
// Wave k computes dot/norm for slot k; then delta + mem_new.
// vo[b,k,m] = v[b,i]*v[b,j] with f = 900k+m, i = f/120, j = f%120.
// ---------------------------------------------------------------------------
__global__ __launch_bounds__(512)
void addressing_kernel(const float* __restrict__ proj, int ldp,
                       const float* __restrict__ memory,
                       float* __restrict__ mem_out,
                       __bf16* __restrict__ A3, int lda3) {
  __shared__ float sva[120], svb[120], sca[16], scb[16];
  __shared__ float smem[900];
  __shared__ float snorm2;

  const int b    = blockIdx.x;
  const int tid  = threadIdx.x;
  const int wave = tid >> 5;
  const int lane = tid & 31;
  const float* pr = proj + (size_t)b * ldp;

  if (tid < 120)       sva[tid]       = pr[tid];
  else if (tid < 240)  svb[tid - 120] = pr[tid];
  if (tid == 0) snorm2 = 0.f;
  __syncthreads();

  float part = 0.f;
  for (int m = tid; m < 900; m += 512) {
    float mv = memory[(size_t)b * MM + m];
    smem[m] = mv;
    part += mv * mv;
  }
#pragma unroll
  for (int off = 16; off; off >>= 1) part += __shfl_xor(part, off, 32);
  if (lane == 0) atomicAdd(&snorm2, part);
  __syncthreads();

  const float mem_norm = fmaxf(sqrtf(snorm2), EPSF);

  {
    const int k = wave;                          // 0..15
    float da = 0.f, na = 0.f, db = 0.f, nb = 0.f;
    for (int m = lane; m < 900; m += 32) {
      int f = 900 * k + m;
      int i = f / 120;
      int j = f - i * 120;
      float pa = sva[i] * sva[j];
      float pb = svb[i] * svb[j];
      float mv = smem[m];
      da += pa * mv; na += pa * pa;
      db += pb * mv; nb += pb * pb;
    }
#pragma unroll
    for (int off = 16; off; off >>= 1) {
      da += __shfl_xor(da, off, 32);
      na += __shfl_xor(na, off, 32);
      db += __shfl_xor(db, off, 32);
      nb += __shfl_xor(nb, off, 32);
    }
    if (lane == 0) {
      float alpha = pr[240 + k];
      float beta  = pr[256 + k];
      float phia  = da / (fmaxf(sqrtf(na), EPSF) * mem_norm);
      float phib  = db / (fmaxf(sqrtf(nb), EPSF) * mem_norm);
      sca[k] = phia * alpha;
      scb[k] = phib * beta;
    }
  }
  __syncthreads();

  for (int m = tid; m < 900; m += 512) {
    float acc = 0.f;
#pragma unroll
    for (int k = 0; k < 16; ++k) {
      int f = 900 * k + m;
      int i = f / 120;
      int j = f - i * 120;
      acc += sca[k] * sva[i] * sva[j] - scb[k] * svb[i] * svb[j];
    }
    float mn = smem[m] + acc;
    mem_out[(size_t)b * MM + m] = mn;
    A3[(size_t)b * lda3 + 512 + m] = (__bf16)mn;
  }
  for (int c = MH + tid; c < KP; c += 512)       // zero pad cols
    A3[(size_t)b * lda3 + c] = (__bf16)0.f;
}

// ---------------------------------------------------------------------------
extern "C" void kernel_launch(void* const* d_in, const int* in_sizes, int n_in,
                              void* d_out, int out_size, void* d_ws, size_t ws_size,
                              hipStream_t stream) {
  (void)in_sizes; (void)n_in; (void)out_size; (void)ws_size;

  const float* x     = (const float*)d_in[0];
  const float* h_in  = (const float*)d_in[1];
  const float* mem   = (const float*)d_in[2];
  const float* W_i2h = (const float*)d_in[3];
  const float* W_h2h = (const float*)d_in[4];
  const float* b_h   = (const float*)d_in[5];
  const float* W_va  = (const float*)d_in[6];
  const float* b_va  = (const float*)d_in[7];
  const float* W_vb  = (const float*)d_in[8];
  const float* b_vb  = (const float*)d_in[9];
  const float* W_a   = (const float*)d_in[10];
  const float* b_a   = (const float*)d_in[11];
  const float* W_b   = (const float*)d_in[12];
  const float* b_b   = (const float*)d_in[13];
  const float* W_o   = (const float*)d_in[14];
  const float* b_o   = (const float*)d_in[15];

  float* out_hout = (float*)d_out;                            // [B,H]
  float* out_h    = out_hout + (size_t)BSZ * H;               // [B,H]
  float* out_mem  = out_h    + (size_t)BSZ * H;               // [B,M]

  // Workspace carve-up (256B aligned)
  char* ws = (char*)d_ws;
  size_t off = 0;
  auto carve = [&](size_t bytes) { char* p = ws + off; off = (off + bytes + 255) & ~(size_t)255; return p; };
  __bf16* A1    = (__bf16*)carve((size_t)BSZ * 1024 * 2);
  __bf16* BT1   = (__bf16*)carve((size_t)512 * 1024 * 2);
  __bf16* CM    = (__bf16*)carve((size_t)BSZ * KP * 2);
  __bf16* BT2   = (__bf16*)carve((size_t)NP * KP * 2);
  float*  PROJ  = (float*) carve((size_t)BSZ * NP * 4);
  float*  BIAS2 = (float*) carve((size_t)NP * 4);
  __bf16* A3    = (__bf16*)carve((size_t)BSZ * KP * 2);
  __bf16* BT3   = (__bf16*)carve((size_t)512 * KP * 2);

  // Stage operands
  pack_A1 <<<(BSZ * 1024 + 255) / 256, 256, 0, stream>>>(x, h_in, A1);
  pack_BT1<<<(512 * 1024 + 255) / 256, 256, 0, stream>>>(W_i2h, W_h2h, BT1);
  pack_CM <<<(BSZ * KP + 255) / 256, 256, 0, stream>>>(h_in, mem, CM);
  pack_BT2<<<(NP * KP + 255) / 256, 256, 0, stream>>>(W_va, W_vb, W_a, W_b, BT2);
  pack_BT3<<<(512 * KP + 255) / 256, 256, 0, stream>>>(W_o, BT3);
  pack_bias2<<<(NP + 63) / 64, 64, 0, stream>>>(b_va, b_vb, b_a, b_b, BIAS2);

  // GEMM1: h = tanh([x,h_prev] @ W1 + b_h) -> out_h (f32) + A3[:,0:512] (bf16)
  gemm_wmma_bf16<true><<<dim3(H / 64, BSZ / 128), 256, 0, stream>>>(
      A1, 1024, BT1, 1024, b_h, out_h, H, A3, KP, 1024);

  // GEMM2: proj = cm @ [W_va;W_vb;W_a;W_b].T + bias -> PROJ
  gemm_wmma_bf16<false><<<dim3(NP / 64, BSZ / 128), 256, 0, stream>>>(
      CM, KP, BT2, KP, BIAS2, PROJ, NP, nullptr, 0, KP);

  // Addressing: mem_new -> out_mem (f32) + A3[:,512:1412] (bf16)
  addressing_kernel<<<BSZ, 512, 0, stream>>>(PROJ, NP, mem, out_mem, A3, KP);

  // GEMM3: h_out = [h, mem_new] @ W_o.T + b_o -> out_hout
  gemm_wmma_bf16<false><<<dim3(H / 64, BSZ / 128), 256, 0, stream>>>(
      A3, KP, BT3, KP, b_o, out_hout, H, nullptr, 0, KP);
}
